// RecurrentCrossAttentionLayer_17171279249471
// MI455X (gfx1250) — compile-verified, matrix operand order checked by reading, not run on hardware
//
#include <hip/hip_runtime.h>

// ---------------------------------------------------------------------------
// RecurrentCrossAttentionLayer (linear attention) for MI455X / gfx1250
// All matmuls via v_wmma_f32_16x16x32_f16 (wave32 WMMA), f32 accumulate.
// State kernel stages K/V tiles with the Tensor Data Mover (TDM).
// ---------------------------------------------------------------------------

typedef __attribute__((ext_vector_type(16))) _Float16 v16h;
typedef __attribute__((ext_vector_type(8)))  _Float16 v8h;
typedef __attribute__((ext_vector_type(4)))  _Float16 v4h;
typedef __attribute__((ext_vector_type(8)))  float    v8f;
typedef __attribute__((ext_vector_type(4)))  float    v4f;
typedef __attribute__((ext_vector_type(4)))  unsigned int v4u;
typedef __attribute__((ext_vector_type(8)))  int      v8i;
typedef __attribute__((ext_vector_type(4)))  int      v4i;

#define D_MODEL 1024
#define N_HEAD  16
#define HEAD_DIM 64
#define SCALING 0.125f   // 64^-0.5
#define EPS 1e-6f

__device__ __forceinline__ v16h cat8(v8h lo, v8h hi) {
  return __builtin_shufflevector(lo, hi, 0,1,2,3,4,5,6,7,8,9,10,11,12,13,14,15);
}
// elu(x)+1  ==  x>0 ? x+1 : exp(x)
__device__ __forceinline__ float feature_map(float x) {
  return x > 0.f ? x + 1.f : __expf(x);
}

// ---------------------------------------------------------------------------
// f32 -> f16 conversion, 4 elements/thread
// ---------------------------------------------------------------------------
__global__ void cvt_f32_f16(const float* __restrict__ src,
                            _Float16* __restrict__ dst, size_t n4) {
  size_t i = (size_t)blockIdx.x * blockDim.x + threadIdx.x;
  if (i >= n4) return;
  v4f s = *(const v4f*)(src + i * 4);
  v4h d;
  d.x = (_Float16)s.x; d.y = (_Float16)s.y;
  d.z = (_Float16)s.z; d.w = (_Float16)s.w;
  *(v4h*)(dst + i * 4) = d;
}

// ---------------------------------------------------------------------------
// Projection GEMM:  P[m,n] = sum_k A[m,k] * W[n,k]  (+ bias[n]), epilogue:
//   MODE 0 (Q): v = fm((acc+b)*SCALING)           -> f16
//   MODE 1 (K): v = fm(acc+b); mask[m] ? 0 : v    -> f16
//   MODE 2 (V): v = acc+b                         -> f16
//   MODE 3 (O): v = acc+b                         -> f32 (final output)
// Block: 256 thr = 8 waves; wave w -> rows [bx*256+w*32,+32), cols [by*64,+64)
// 2 A-fragments reuse each B-fragment (halves weight traffic vs 16-row tile).
// Whole f16 working set (~110MB) is resident in the 192MB L2, so fragments
// are fetched straight from global.
// ---------------------------------------------------------------------------
template <int MODE>
__global__ __launch_bounds__(256) void gemm_proj(
    const _Float16* __restrict__ A,    // [M x K] row-major f16
    const _Float16* __restrict__ W,    // [N x K] row-major f16 (B = W^T)
    const float*    __restrict__ bias, // [N]
    const unsigned char* __restrict__ mask, // [M] (MODE 1 only)
    _Float16* __restrict__ out16,
    float*    __restrict__ out32,
    int M, int N, int K) {
  const int lane    = threadIdx.x & 31;
  const int wave    = threadIdx.x >> 5;
  const int rowBase = blockIdx.x * 256 + wave * 32;
  const int colBase = blockIdx.y * 64;
  const int half16  = (lane < 16) ? 0 : 8;   // K-chunk select / M-row select
  const int lr      = lane & 15;

  v8f acc[2][4] = {};
  const _Float16* aPtr0 = A + (size_t)(rowBase + lr) * K;
  const _Float16* aPtr1 = aPtr0 + (size_t)16 * K;

  for (int k = 0; k < K; k += 32) {
    const int k0 = k + half16;
    __builtin_prefetch(aPtr0 + k + 64, 0, 0);         // global_prefetch_b8
    v16h afrag0 = cat8(*(const v8h*)(aPtr0 + k0),
                       *(const v8h*)(aPtr0 + k0 + 16));
    v16h afrag1 = cat8(*(const v8h*)(aPtr1 + k0),
                       *(const v8h*)(aPtr1 + k0 + 16));
#pragma unroll
    for (int t = 0; t < 4; ++t) {
      const _Float16* wPtr = W + (size_t)(colBase + t * 16 + lr) * K + k0;
      v16h bfrag = cat8(*(const v8h*)(wPtr),
                        *(const v8h*)(wPtr + 16));
      acc[0][t] = __builtin_amdgcn_wmma_f32_16x16x32_f16(
          false, afrag0, false, bfrag, (short)0, acc[0][t], false, false);
      acc[1][t] = __builtin_amdgcn_wmma_f32_16x16x32_f16(
          false, afrag1, false, bfrag, (short)0, acc[1][t], false, false);
    }
  }
#pragma unroll
  for (int i = 0; i < 2; ++i) {
#pragma unroll
    for (int t = 0; t < 4; ++t) {
#pragma unroll
      for (int r = 0; r < 8; ++r) {
        const int m = rowBase + i * 16 + half16 + r;
        const int n = colBase + t * 16 + lr;
        float v = acc[i][t][r] + bias[n];
        if (MODE == 0) { v = feature_map(v * SCALING); }
        if (MODE == 1) { v = feature_map(v); if (mask[m]) v = 0.f; }
        if (MODE == 3) out32[(size_t)m * N + n] = v;
        else           out16[(size_t)m * N + n] = (_Float16)v;
      }
    }
  }
}

// ---------------------------------------------------------------------------
// State: S[b,h,m,d] = sum_s K[b,s,h,d] * V[b,s,h,m]   (64x64 per head)
// One block per (b,h); 512 thr = 16 waves in 4x4 tile grid.
// K/V tiles (32s x 64) are DMA'd into LDS by the Tensor Data Mover (wave 0
// issues TENSOR_LOAD_TO_LDS, completes with s_wait_tensorcnt), then threads
// transpose 16-bit elements LDS->LDS so both WMMA operands read contiguous
// K-chunks (A = V^T rows, B = K columns) via ds_load_b128.
// ---------------------------------------------------------------------------
__global__ __launch_bounds__(512) void state_kernel(
    const _Float16* __restrict__ Kf,  // [B*Lk, D_MODEL]
    const _Float16* __restrict__ Vf,  // [B*Lk, D_MODEL]
    _Float16* __restrict__ S16,       // [B*H, 64, 64] (m-major, d inner)
    int Lk) {
  __shared__ __align__(16) _Float16 KNAT[32 * 64];  // natural [s][d]
  __shared__ __align__(16) _Float16 VNAT[32 * 64];
  __shared__ __align__(16) _Float16 KT[64 * 40];    // [d][s] stride 40
  __shared__ __align__(16) _Float16 VT[64 * 40];    // [m][s] stride 40
  const int bh = blockIdx.x;
  const int b = bh >> 4, h = bh & 15;
  const int lane = threadIdx.x & 31;
  const int wave = threadIdx.x >> 5;
  const int wm = (wave >> 2) * 16;   // m tile
  const int wd = (wave & 3) * 16;    // d tile
  const int half16 = (lane < 16) ? 0 : 8;
  const int lr = lane & 15;

  v8f acc = {};
  const size_t base = (size_t)b * Lk * D_MODEL + h * HEAD_DIM;
  const unsigned long long gK = (unsigned long long)(size_t)(Kf + base);
  const unsigned long long gV = (unsigned long long)(size_t)(Vf + base);
  // LDS generic pointers carry the LDS byte offset in the low 32 bits.
  const unsigned int ldsK = (unsigned int)(size_t)&KNAT[0];
  const unsigned int ldsV = (unsigned int)(size_t)&VNAT[0];

  // D# group 1: workgroup_mask=0, data_size=2B, no barrier/iterate/pad;
  // tensor_dim0=64, tensor_dim1=Lk, tile=64x32, dim0_stride=D_MODEL elems.
  v8i g1;
  g1[0] = 0x00010000;
  g1[1] = (64 << 16);
  g1[2] = (int)(((unsigned)Lk & 0xFFFFu) << 16);
  g1[3] = (int)((((unsigned)Lk >> 16) & 0xFFFFu) | (64u << 16));
  g1[4] = 32;
  g1[5] = D_MODEL;
  g1[6] = 0;
  g1[7] = 0;
  const v4i gz4 = {0, 0, 0, 0};
  const v8i gz8 = {0, 0, 0, 0, 0, 0, 0, 0};

  const int e = threadIdx.x * 4;           // 512 thr * 4 = 2048 = 32x64 tile
  const int ts = e >> 6, td = e & 63;      // tile row s, col d (d mult of 4)

  for (int s0 = 0; s0 < Lk; s0 += 32) {
    __syncthreads();                       // protect KT/VT reads of prev iter
    if (wave == 0) {
      const unsigned long long gaK = gK + (unsigned long long)s0 * (D_MODEL * 2);
      const unsigned long long gaV = gV + (unsigned long long)s0 * (D_MODEL * 2);
      // D# group 0: count=1, lds_addr, global_addr[56:0], type=2
      v4u g0k = { 1u, ldsK, (unsigned int)gaK,
                  (unsigned int)((gaK >> 32) & 0x1FFFFFFu) | (2u << 30) };
      v4u g0v = { 1u, ldsV, (unsigned int)gaV,
                  (unsigned int)((gaV >> 32) & 0x1FFFFFFu) | (2u << 30) };
      __builtin_amdgcn_tensor_load_to_lds(g0k, g1, gz4, gz4, gz8, 0);
      __builtin_amdgcn_tensor_load_to_lds(g0v, g1, gz4, gz4, gz8, 0);
      __builtin_amdgcn_s_wait_tensorcnt(0);
    }
    __syncthreads();                       // tiles now in KNAT/VNAT
    {
      v4h kv = *(const v4h*)&KNAT[ts * 64 + td];
      v4h vv = *(const v4h*)&VNAT[ts * 64 + td];
#pragma unroll
      for (int j = 0; j < 4; ++j) {
        KT[(td + j) * 40 + ts] = kv[j];
        VT[(td + j) * 40 + ts] = vv[j];
      }
    }
    __syncthreads();                       // transposed tiles ready
    const int k0 = half16;
    v16h afrag = cat8(*(const v8h*)&VT[(wm + lr) * 40 + k0],
                      *(const v8h*)&VT[(wm + lr) * 40 + k0 + 16]);
    v16h bfrag = cat8(*(const v8h*)&KT[(wd + lr) * 40 + k0],
                      *(const v8h*)&KT[(wd + lr) * 40 + k0 + 16]);
    acc = __builtin_amdgcn_wmma_f32_16x16x32_f16(
        false, afrag, false, bfrag, (short)0, acc, false, false);
  }
#pragma unroll
  for (int r = 0; r < 8; ++r) {
    const int m = wm + half16 + r;
    const int d = wd + lr;
    S16[(size_t)bh * 4096 + m * 64 + d] = (_Float16)acc[r];
  }
}

// ---------------------------------------------------------------------------
// Z[b,h,d] = sum_s K[b,s,h,d]
// ---------------------------------------------------------------------------
__global__ __launch_bounds__(256) void zsum_kernel(
    const _Float16* __restrict__ Kf, float* __restrict__ Z, int Lk) {
  __shared__ float sz[256];
  const int bh = blockIdx.x;
  const int b = bh >> 4, h = bh & 15;
  const int d = threadIdx.x & 63;
  const int j = threadIdx.x >> 6;  // 0..3
  const _Float16* kp = Kf + (size_t)b * Lk * D_MODEL + h * HEAD_DIM + d;
  float acc = 0.f;
  for (int s = j; s < Lk; s += 4) acc += (float)kp[(size_t)s * D_MODEL];
  sz[threadIdx.x] = acc;
  __syncthreads();
  if (j == 0) Z[bh * 64 + d] = sz[d] + sz[64 + d] + sz[128 + d] + sz[192 + d];
}

// ---------------------------------------------------------------------------
// qz[b,q,h] = 1 / (sum_d Q[b,q,h,d] * Z[b,h,d] + eps)
// ---------------------------------------------------------------------------
__global__ void qz_kernel(const _Float16* __restrict__ Qf,
                          const float* __restrict__ Z,
                          float* __restrict__ qz, int total, int Lq) {
  const int i = blockIdx.x * blockDim.x + threadIdx.x;
  if (i >= total) return;
  const int h = i & 15;
  const int bq = i >> 4;               // b*Lq + q
  const int b = bq / Lq;
  const _Float16* qp = Qf + (size_t)bq * D_MODEL + h * HEAD_DIM;
  const float* zp = Z + (b * 16 + h) * 64;
  float acc = 0.f;
#pragma unroll 8
  for (int d = 0; d < 64; ++d) acc += (float)qp[d] * zp[d];
  qz[i] = 1.0f / (acc + EPS);
}

// ---------------------------------------------------------------------------
// Out[b,q,h,m] = qz[b,q,h] * sum_d Q[b,q,h,d] * S[b,h,m,d]
// Block: 256 thr = 8 waves; wave -> 16 q-rows x 64 m-cols; K = 64 (2 steps).
// B operand: lane = column m, contiguous d-chunks of S[m,:].
// ---------------------------------------------------------------------------
__global__ __launch_bounds__(256) void attn_out_kernel(
    const _Float16* __restrict__ Qf, const _Float16* __restrict__ S16,
    const float* __restrict__ qz, _Float16* __restrict__ hid, int Lq) {
  const int bh = blockIdx.y;
  const int b = bh >> 4, h = bh & 15;
  const int lane = threadIdx.x & 31;
  const int wave = threadIdx.x >> 5;
  const int qBase = blockIdx.x * 128 + wave * 16;
  const int half16 = (lane < 16) ? 0 : 8;
  const int lr = lane & 15;

  v8f acc[4] = {};
  const _Float16* qp = Qf + (size_t)(b * Lq + qBase + lr) * D_MODEL + h * HEAD_DIM;
  const _Float16* sp = S16 + (size_t)bh * 4096;

#pragma unroll
  for (int kk = 0; kk < 2; ++kk) {
    const int k0 = kk * 32 + half16;
    v16h afrag = cat8(*(const v8h*)(qp + k0), *(const v8h*)(qp + k0 + 16));
#pragma unroll
    for (int t = 0; t < 4; ++t) {
      const _Float16* bp = sp + (t * 16 + lr) * 64 + k0;
      v16h bfrag = cat8(*(const v8h*)bp, *(const v8h*)(bp + 16));
      acc[t] = __builtin_amdgcn_wmma_f32_16x16x32_f16(
          false, afrag, false, bfrag, (short)0, acc[t], false, false);
    }
  }
#pragma unroll
  for (int r = 0; r < 8; ++r) {
    const int q = qBase + half16 + r;
    const float s = qz[(b * Lq + q) * 16 + h];
#pragma unroll
    for (int t = 0; t < 4; ++t) {
      const int m = t * 16 + lr;
      hid[(size_t)(b * Lq + q) * D_MODEL + h * HEAD_DIM + m] =
          (_Float16)(acc[t][r] * s);
    }
  }
}

// ---------------------------------------------------------------------------
// Host-side launch
// ---------------------------------------------------------------------------
extern "C" void kernel_launch(void* const* d_in, const int* in_sizes, int n_in,
                              void* d_out, int out_size, void* d_ws, size_t ws_size,
                              hipStream_t stream) {
  (void)in_sizes; (void)n_in; (void)out_size; (void)ws_size;
  const int B = 4, L = 2048;
  const int M = B * L;                 // 8192 rows
  const size_t actElems = (size_t)M * D_MODEL;      // 8.39M
  const size_t wElems   = (size_t)D_MODEL * D_MODEL;

  const float* querys = (const float*)d_in[0];
  const float* keys   = (const float*)d_in[1];
  const float* values = (const float*)d_in[2];
  const unsigned char* key_mask = (const unsigned char*)d_in[3];
  const float* Wq = (const float*)d_in[4];  const float* bq = (const float*)d_in[5];
  const float* Wk = (const float*)d_in[6];  const float* bk = (const float*)d_in[7];
  const float* Wv = (const float*)d_in[8];  const float* bv = (const float*)d_in[9];
  const float* Wo = (const float*)d_in[10]; const float* bo = (const float*)d_in[11];
  float* out = (float*)d_out;

  // workspace carve-up (256B aligned offsets)
  char* ws = (char*)d_ws;
  size_t off = 0;
  auto carve = [&](size_t bytes) {
    char* p = ws + off;
    off += (bytes + 255) & ~(size_t)255;
    return p;
  };
  _Float16* xq16 = (_Float16*)carve(actElems * 2);
  _Float16* xk16 = (_Float16*)carve(actElems * 2);
  _Float16* xv16 = (_Float16*)carve(actElems * 2);
  _Float16* Wq16 = (_Float16*)carve(wElems * 2);
  _Float16* Wk16 = (_Float16*)carve(wElems * 2);
  _Float16* Wv16 = (_Float16*)carve(wElems * 2);
  _Float16* Wo16 = (_Float16*)carve(wElems * 2);
  _Float16* Qf   = (_Float16*)carve(actElems * 2);
  _Float16* Kf   = (_Float16*)carve(actElems * 2);
  _Float16* Vf   = (_Float16*)carve(actElems * 2);
  _Float16* S16  = (_Float16*)carve((size_t)B * N_HEAD * 64 * 64 * 2);
  float*    Zb   = (float*)carve((size_t)B * N_HEAD * 64 * 4);
  float*    qz   = (float*)carve((size_t)B * L * N_HEAD * 4);
  _Float16* hid  = xq16;  // xq16 dead after Q projection -> reuse for hidden

  // 1) f32 -> f16 conversions
  {
    const size_t n4a = actElems / 4, n4w = wElems / 4;
    const int blk = 256;
    cvt_f32_f16<<<dim3((unsigned)((n4a + blk - 1) / blk)), blk, 0, stream>>>(querys, xq16, n4a);
    cvt_f32_f16<<<dim3((unsigned)((n4a + blk - 1) / blk)), blk, 0, stream>>>(keys,   xk16, n4a);
    cvt_f32_f16<<<dim3((unsigned)((n4a + blk - 1) / blk)), blk, 0, stream>>>(values, xv16, n4a);
    cvt_f32_f16<<<dim3((unsigned)((n4w + blk - 1) / blk)), blk, 0, stream>>>(Wq, Wq16, n4w);
    cvt_f32_f16<<<dim3((unsigned)((n4w + blk - 1) / blk)), blk, 0, stream>>>(Wk, Wk16, n4w);
    cvt_f32_f16<<<dim3((unsigned)((n4w + blk - 1) / blk)), blk, 0, stream>>>(Wv, Wv16, n4w);
    cvt_f32_f16<<<dim3((unsigned)((n4w + blk - 1) / blk)), blk, 0, stream>>>(Wo, Wo16, n4w);
  }

  // 2) projections (WMMA): grid (M/256, N/64)
  dim3 gGemm(M / 256, D_MODEL / 64);
  gemm_proj<0><<<gGemm, 256, 0, stream>>>(xq16, Wq16, bq, nullptr, Qf, nullptr, M, D_MODEL, D_MODEL);
  gemm_proj<1><<<gGemm, 256, 0, stream>>>(xk16, Wk16, bk, key_mask, Kf, nullptr, M, D_MODEL, D_MODEL);
  gemm_proj<2><<<gGemm, 256, 0, stream>>>(xv16, Wv16, bv, nullptr, Vf, nullptr, M, D_MODEL, D_MODEL);

  // 3) Z = sum_s K ; S = V^T K (per head, WMMA + TDM tile staging)
  zsum_kernel<<<B * N_HEAD, 256, 0, stream>>>(Kf, Zb, L);
  state_kernel<<<B * N_HEAD, 512, 0, stream>>>(Kf, Vf, S16, L);

  // 4) qz = 1/(Q.Z + eps)
  {
    const int total = B * L * N_HEAD;
    qz_kernel<<<(total + 255) / 256, 256, 0, stream>>>(Qf, Zb, qz, total, L);
  }

  // 5) attention output -> hid (f16)
  attn_out_kernel<<<dim3(L / 128, B * N_HEAD), 256, 0, stream>>>(Qf, S16, qz, hid, L);

  // 6) final projection with Wo (+bo) -> f32 out
  gemm_proj<3><<<gGemm, 256, 0, stream>>>(hid, Wo16, bo, nullptr, nullptr, out, M, D_MODEL, D_MODEL);
}
